// MambaController_53386443490000
// MI455X (gfx1250) — matrix operand order repaired and checked by reading, not compiled, not run on hardware
//
#include <hip/hip_runtime.h>

typedef __attribute__((ext_vector_type(16))) __bf16 v16bf;
typedef __attribute__((ext_vector_type(8)))  __bf16 v8bf;
typedef __attribute__((ext_vector_type(8)))  float  v8f;

namespace {
constexpr int BB = 8, LL = 2048, OBS = 64, ACT = 16;
constexpr int DM = 768, DI = 1536, NST = 16, RR = 48, NL = 2;
constexpr int MT = BB * LL;  // 16384 token rows
}

__device__ __forceinline__ float siluf(float x) { return x / (1.f + __expf(-x)); }
__device__ __forceinline__ float softplusf(float x) {
  return fmaxf(x, 0.f) + log1pf(__expf(-fabsf(x)));
}

// ---------------- cast + zero-pad fp32 -> bf16 -----------------------------
__global__ void k_cast_pad(const float* __restrict__ src, __bf16* __restrict__ dst,
                           int rows, int cols, int src_ld, int drows, int dcols) {
  size_t i = (size_t)blockIdx.x * blockDim.x + threadIdx.x;
  size_t total = (size_t)drows * dcols;
  if (i >= total) return;
  int r = (int)(i / dcols), c = (int)(i % dcols);
  float v = (r < rows && c < cols) ? src[(size_t)r * src_ld + c] : 0.f;
  dst[i] = (__bf16)v;
}

// ---------------- WMMA bf16 GEMM: C[M,N] = A[M,K] * W[N,K]^T ---------------
// Async-copy double-buffered LDS pipeline (ASYNCcnt), block tile 128x64,
// wave tile 32x64 (2 A-frags x 4 B-frags = 8 WMMA per k-step per wave).
// K multiple of 32; weight rows zero-padded to multiple of 64; M mult of 128.
// EPI: 0 = none, 1 = +bias, 2 = +bias then softplus
template <int EPI>
__global__ void __launch_bounds__(128)
k_gemm_wmma(const __bf16* __restrict__ A, const __bf16* __restrict__ W,
            const float* __restrict__ bias, float* __restrict__ Cf,
            __bf16* __restrict__ Cbf, int K, int N, int ldc) {
  __shared__ __align__(16) __bf16 sA[2][128 * 32];  // 16 KB
  __shared__ __align__(16) __bf16 sB[2][64 * 32];   //  8 KB
  const int tid  = threadIdx.x;        // 0..127
  const int lane = tid & 31;
  const int wave = tid >> 5;           // wave32
  const int mblk = blockIdx.x * 128;
  const int nblk = blockIdx.y * 64;
  const int hv   = lane >> 4;
  const int l15  = lane & 15;

  // issue 6 async b128 copies per thread-slot: A tile 128x32, B tile 64x32
  auto stage = [&](int s, int k0) {
#pragma unroll
    for (int q = 0; q < 4; ++q) {                  // A: 512 16B chunks / 128 thr
      int chunk = q * 128 + tid;
      int row = chunk >> 2, seg = chunk & 3;       // 4 chunks per 32-elem row
      const __bf16* g = A + (size_t)(mblk + row) * K + k0 + seg * 8;
      unsigned laddr = (unsigned)(size_t)&sA[s][row * 32 + seg * 8];
      asm volatile("global_load_async_to_lds_b128 %0, %1, off"
                   :: "v"(laddr), "v"((unsigned long long)(size_t)g) : "memory");
    }
#pragma unroll
    for (int q = 0; q < 2; ++q) {                  // B: 256 16B chunks / 128 thr
      int chunk = q * 128 + tid;
      int row = chunk >> 2, seg = chunk & 3;
      const __bf16* g = W + (size_t)(nblk + row) * K + k0 + seg * 8;
      unsigned laddr = (unsigned)(size_t)&sB[s][row * 32 + seg * 8];
      asm volatile("global_load_async_to_lds_b128 %0, %1, off"
                   :: "v"(laddr), "v"((unsigned long long)(size_t)g) : "memory");
    }
  };

  v8f acc[2][4] = {{v8f{}, v8f{}, v8f{}, v8f{}}, {v8f{}, v8f{}, v8f{}, v8f{}}};

  stage(0, 0);
  int s = 0;
  for (int k0 = 0; k0 < K; k0 += 32, s ^= 1) {
    if (k0 + 32 < K) {
      stage(s ^ 1, k0 + 32);                       // prefetch next slab
      asm volatile("s_wait_asynccnt 6" ::: "memory");   // slab s done, next in flight
    } else {
      asm volatile("s_wait_asynccnt 0" ::: "memory");
    }
    __syncthreads();

    // A fragments from LDS: row = local M; e<8 -> K=hv*8+e ; e>=8 -> 16+hv*8+(e-8)
    v16bf af[2];
#pragma unroll
    for (int i = 0; i < 2; ++i) {
      const __bf16* ap = &sA[s][(wave * 32 + i * 16 + l15) * 32 + hv * 8];
      v8bf alo = *(const v8bf*)(ap);
      v8bf ahi = *(const v8bf*)(ap + 16);
#pragma unroll
      for (int e = 0; e < 8; ++e) { af[i][e] = alo[e]; af[i][e + 8] = ahi[e]; }
    }
#pragma unroll
    for (int j = 0; j < 4; ++j) {
      // B fragment: lane holds col N=l15; e<8 -> K=hv*8+e ; e>=8 -> 16+hv*8+(e-8)
      const __bf16* bp = &sB[s][(j * 16 + l15) * 32 + hv * 8];
      v8bf blo = *(const v8bf*)(bp);
      v8bf bhi = *(const v8bf*)(bp + 16);
      v16bf bf;
#pragma unroll
      for (int e = 0; e < 8; ++e) { bf[e] = blo[e]; bf[e + 8] = bhi[e]; }
#pragma unroll
      for (int i = 0; i < 2; ++i)
        acc[i][j] = __builtin_amdgcn_wmma_f32_16x16x32_bf16(
            false, af[i], false, bf, (short)0, acc[i][j], false, false);
    }
    __syncthreads();  // all waves done reading slot s before it is re-staged
  }

  // store: m = mblk + wave*32 + i*16 + hv*8 + r ; n = nblk + j*16 + l15
#pragma unroll
  for (int i = 0; i < 2; ++i) {
#pragma unroll
    for (int j = 0; j < 4; ++j) {
      const int n = nblk + j * 16 + l15;
      if (n >= N) continue;
#pragma unroll
      for (int r = 0; r < 8; ++r) {
        const int m = mblk + wave * 32 + i * 16 + hv * 8 + r;
        float v = acc[i][j][r];
        if (EPI >= 1) v += bias[n];
        if (EPI == 2) v = softplusf(v);
        if (Cf)  Cf[(size_t)m * ldc + n] = v;
        if (Cbf) Cbf[(size_t)m * ldc + n] = (__bf16)v;
      }
    }
  }
}

// ---------------- causal depthwise conv (K=4) + bias + silu ----------------
__global__ void k_conv_silu(const float* __restrict__ xz, const float* __restrict__ cw,
                            const float* __restrict__ cb, float* __restrict__ xaf,
                            __bf16* __restrict__ xab) {
  size_t i = (size_t)blockIdx.x * blockDim.x + threadIdx.x;
  if (i >= (size_t)MT * DI) return;
  int d = (int)(i % DI);
  size_t row = i / DI;
  int l = (int)(row % LL);
  float acc = cb[d];
#pragma unroll
  for (int k = 0; k < 4; ++k) {
    int ll = l - 3 + k;
    if (ll >= 0) acc += cw[d * 4 + k] * xz[(row - 3 + (size_t)k) * (2 * DI) + d];
  }
  float v = siluf(acc);
  xaf[i] = v;
  xab[i] = (__bf16)v;
}

// ---------------- selective scan, fused D-skip + silu(z) gate --------------
__global__ void k_scan(const float* __restrict__ xa, const float* __restrict__ dt,
                       const float* __restrict__ xdbl, const float* __restrict__ xz,
                       const float* __restrict__ alog, const float* __restrict__ dsk,
                       __bf16* __restrict__ ybf) {
  __shared__ float sBC[32];
  int tid = blockIdx.x * 256 + threadIdx.x;  // 0 .. B*DI-1
  int b = tid / DI, d = tid % DI;            // DI = 6*256 -> whole block same b

  float Ac[NST];
#pragma unroll
  for (int n = 0; n < NST; ++n) Ac[n] = -__expf(alog[d * NST + n]);
  const float Dv = dsk[d];

  float h[NST];
#pragma unroll
  for (int n = 0; n < NST; ++n) h[n] = 0.f;

  for (int l = 0; l < LL; ++l) {
    size_t row = (size_t)b * LL + l;
    if (threadIdx.x < 32) sBC[threadIdx.x] = xdbl[row * (RR + 2 * NST) + RR + threadIdx.x];
    __syncthreads();

    float xv  = xa[row * DI + d];
    float dtv = dt[row * DI + d];
    float zv  = xz[row * (2 * DI) + DI + d];
    float dx  = dtv * xv;
    float y   = 0.f;
#pragma unroll
    for (int n = 0; n < NST; ++n) {
      float dA = __expf(dtv * Ac[n]);
      h[n] = dA * h[n] + dx * sBC[n];
      y += h[n] * sBC[16 + n];
    }
    y += xv * Dv;
    ybf[row * DI + d] = (__bf16)(y * siluf(zv));
    __syncthreads();
  }
}

// ---------------------------------------------------------------------------
extern "C" void kernel_launch(void* const* d_in, const int* in_sizes, int n_in,
                              void* d_out, int out_size, void* d_ws, size_t ws_size,
                              hipStream_t stream) {
  const float* obs  = (const float*)d_in[0];
  const float* w_in = (const float*)d_in[1];
  const float* b_in = (const float*)d_in[2];
  const float* ipw  = (const float*)d_in[3];
  const float* cw   = (const float*)d_in[4];
  const float* cb   = (const float*)d_in[5];
  const float* xpw  = (const float*)d_in[6];
  const float* dpw  = (const float*)d_in[7];
  const float* dpb  = (const float*)d_in[8];
  const float* alog = (const float*)d_in[9];
  const float* dsk  = (const float*)d_in[10];
  const float* opw  = (const float*)d_in[11];
  const float* wout = (const float*)d_in[12];
  const float* bout = (const float*)d_in[13];
  float* out = (float*)d_out;
  (void)in_sizes; (void)n_in; (void)out_size; (void)ws_size;

  char* p = (char*)d_ws;
  auto alloc = [&](size_t bytes) -> char* {
    char* q = p; p += (bytes + 255) & ~(size_t)255; return q;
  };
  __bf16* obs_bf = (__bf16*)alloc((size_t)MT * OBS * 2);
  __bf16* win_bf = (__bf16*)alloc((size_t)DM * OBS * 2);
  __bf16* x_bf   = (__bf16*)alloc((size_t)MT * DM * 2);
  __bf16* ipw_bf = (__bf16*)alloc((size_t)(2 * DI) * DM * 2);
  float*  xz_f   = (float*)alloc((size_t)MT * 2 * DI * 4);
  float*  xa_f   = (float*)alloc((size_t)MT * DI * 4);
  __bf16* xa_bf  = (__bf16*)alloc((size_t)MT * DI * 2);
  __bf16* xpw_bf = (__bf16*)alloc((size_t)128 * DI * 2);   // 80 -> 128 rows
  float*  xdbl_f = (float*)alloc((size_t)MT * (RR + 2 * NST) * 4);
  __bf16* dtr_bf = (__bf16*)alloc((size_t)MT * 64 * 2);    // K 48 -> 64
  __bf16* dpw_bf = (__bf16*)alloc((size_t)DI * 64 * 2);
  float*  dt_f   = (float*)alloc((size_t)MT * DI * 4);
  __bf16* y_bf   = (__bf16*)alloc((size_t)MT * DI * 2);
  __bf16* opw_bf = (__bf16*)alloc((size_t)DM * DI * 2);
  __bf16* wou_bf = (__bf16*)alloc((size_t)64 * DM * 2);    // 16 -> 64 rows

  auto cast = [&](const float* s, __bf16* d, int rows, int cols, int sld,
                  int drows, int dcols) {
    size_t tot = (size_t)drows * dcols;
    k_cast_pad<<<dim3((unsigned)((tot + 255) / 256)), dim3(256), 0, stream>>>(
        s, d, rows, cols, sld, drows, dcols);
  };
  const dim3 gblk(128);

  // ---- input projection: x = obs @ w_in^T + b_in  (bf16 out) ----
  cast(obs, obs_bf, MT, OBS, OBS, MT, OBS);
  cast(w_in, win_bf, DM, OBS, OBS, DM, OBS);
  k_gemm_wmma<1><<<dim3(MT / 128, DM / 64), gblk, 0, stream>>>(
      obs_bf, win_bf, b_in, nullptr, x_bf, OBS, DM, DM);

  for (int i = 0; i < NL; ++i) {
    const float* ipw_i = ipw + (size_t)i * 2 * DI * DM;
    const float* cw_i  = cw  + (size_t)i * DI * 4;
    const float* cb_i  = cb  + (size_t)i * DI;
    const float* xpw_i = xpw + (size_t)i * (RR + 2 * NST) * DI;
    const float* dpw_i = dpw + (size_t)i * DI * RR;
    const float* dpb_i = dpb + (size_t)i * DI;
    const float* al_i  = alog + (size_t)i * DI * NST;
    const float* ds_i  = dsk + (size_t)i * DI;
    const float* opw_i = opw + (size_t)i * DM * DI;

    // in_proj: xz = x @ in_proj_w^T   [MT, 3072] f32
    cast(ipw_i, ipw_bf, 2 * DI, DM, DM, 2 * DI, DM);
    k_gemm_wmma<0><<<dim3(MT / 128, (2 * DI) / 64), gblk, 0, stream>>>(
        x_bf, ipw_bf, nullptr, xz_f, nullptr, DM, 2 * DI, 2 * DI);

    // causal depthwise conv + silu
    k_conv_silu<<<dim3((unsigned)(((size_t)MT * DI + 255) / 256)), dim3(256), 0, stream>>>(
        xz_f, cw_i, cb_i, xa_f, xa_bf);

    // x_proj: x_dbl = xa @ x_proj_w^T   [MT, 80] f32
    cast(xpw_i, xpw_bf, RR + 2 * NST, DI, DI, 128, DI);
    k_gemm_wmma<0><<<dim3(MT / 128, 2), gblk, 0, stream>>>(
        xa_bf, xpw_bf, nullptr, xdbl_f, nullptr, DI, RR + 2 * NST, RR + 2 * NST);

    // dt_proj: dt = softplus(dt_r @ dt_w^T + dt_b)   [MT, 1536] f32
    cast(xdbl_f, dtr_bf, MT, RR, RR + 2 * NST, MT, 64);
    cast(dpw_i, dpw_bf, DI, RR, RR, DI, 64);
    k_gemm_wmma<2><<<dim3(MT / 128, DI / 64), gblk, 0, stream>>>(
        dtr_bf, dpw_bf, dpb_i, dt_f, nullptr, 64, DI, DI);

    // selective scan (+D skip, *silu(z)) -> y_bf
    k_scan<<<dim3((BB * DI) / 256), dim3(256), 0, stream>>>(
        xa_f, dt_f, xdbl_f, xz_f, al_i, ds_i, y_bf);

    // out_proj: x = y @ out_proj_w^T   (bf16 out, becomes next layer input)
    cast(opw_i, opw_bf, DM, DI, DI, DM, DI);
    k_gemm_wmma<0><<<dim3(MT / 128, DM / 64), gblk, 0, stream>>>(
        y_bf, opw_bf, nullptr, nullptr, x_bf, DI, DM, DM);
  }

  // ---- output head: out = x @ w_out^T + b_out   [MT, 16] f32 ----
  cast(wout, wou_bf, ACT, DM, DM, 64, DM);
  k_gemm_wmma<1><<<dim3(MT / 128, 1), gblk, 0, stream>>>(
      x_bf, wou_bf, bout, out, nullptr, DM, ACT, ACT);
}
